// MatchModule_68676527063223
// MI455X (gfx1250) — compile-verified, matrix operand order checked by reading, not run on hardware
//
#include <hip/hip_runtime.h>

// ---------------- problem dims ----------------
#define Hd    128
#define HEADS 4
#define DKd   32
#define Bd    8
#define Kd_   256      // K objects
#define Ld    8
#define CDET  1152
#define CCLIP 512
#define NFd   4
#define NWd   64
#define BLd   64       // B*L

typedef __bf16 v16bf __attribute__((ext_vector_type(16)));
typedef float  v8f   __attribute__((ext_vector_type(8)));

__device__ __forceinline__ int kmap(int i, int kh) {
    // 16-bit A/B fragment K mapping for V_WMMA_*_16X16X32 (wave32):
    // lanes 0-15 hold K pairs {0..7},{16..23}; lanes 16-31 hold {8..15},{24..31}
    return ((i < 4) ? (2 * i) : (16 + 2 * (i - 4))) + 8 * kh;
}

__device__ __forceinline__ v8f wmma_bf16(v16bf a, v16bf b, v8f c) {
    return __builtin_amdgcn_wmma_f32_16x16x32_bf16(false, a, false, b, (short)0, c, false, false);
}

// ---------------- weight convert: fp32 [K][N] -> bf16 [N][K] ----------------
__global__ void k_convT(const float* __restrict__ src, __bf16* __restrict__ dst, int Kdim, int N) {
    int idx = blockIdx.x * blockDim.x + threadIdx.x;
    if (idx >= Kdim * N) return;
    int n = idx / Kdim, k = idx % Kdim;
    dst[idx] = (__bf16)src[(size_t)k * N + n];
}

// ---------------- distance-bias: dwgt[b][0][i][j]=w/sum, [b][1][i][j]=-dist ----------------
__global__ void k_dist(const float* __restrict__ center, float* __restrict__ dwgt) {
    int b = blockIdx.x / Kd_, i = blockIdx.x % Kd_, j = threadIdx.x;
    const float* cb = center + (size_t)b * Kd_ * 3;
    float dx = cb[i * 3 + 0] - cb[j * 3 + 0];
    float dy = cb[i * 3 + 1] - cb[j * 3 + 1];
    float dz = cb[i * 3 + 2] - cb[j * 3 + 2];
    float dist = sqrtf(dx * dx + dy * dy + dz * dz);
    float w = 1.0f / (dist + 0.01f);
    __shared__ float red[Kd_];
    red[j] = w;
    __syncthreads();
    for (int s = Kd_ / 2; s > 0; s >>= 1) {
        if (j < s) red[j] += red[j + s];
        __syncthreads();
    }
    float invs = 1.0f / red[0];
    size_t base = ((size_t)b * 2) * Kd_ * Kd_ + (size_t)i * Kd_ + j;
    dwgt[base] = w * invs;
    dwgt[base + (size_t)Kd_ * Kd_] = -dist;
}

// ---------------- generic WMMA GEMM: C[M][128] = epi((A[+A2]) @ Bt) ----------------
// Bt: bf16 transposed weights [128][Kdim]. One wave -> 16x64 tile.
// Template flags keep runtime pointer tests out of the hot loop (no exec-mask churn).
template <bool HAS_A2, bool HAS_BNPRELU, bool ALPHA_SCALAR>
__global__ __launch_bounds__(256)
void k_gemm(const float* __restrict__ A, const float* __restrict__ A2,
            const __bf16* __restrict__ Bt, float* __restrict__ Cout,
            int M, int Kdim,
            const float* __restrict__ bias, const float* __restrict__ g,
            const float* __restrict__ bb, const float* __restrict__ alpha)
{
    int wave = (blockIdx.x * blockDim.x + threadIdx.x) >> 5;
    int lane = threadIdx.x & 31;
    int row = (wave >> 1) * 16;      // 2 column tiles of 64
    int col = (wave & 1) * 64;
    if (row >= M) return;

    int m = lane & 15, kh = lane >> 4;
    v8f acc[4];
    acc[0] = (v8f){}; acc[1] = (v8f){}; acc[2] = (v8f){}; acc[3] = (v8f){};

    for (int k0 = 0; k0 < Kdim; k0 += 32) {
        v16bf a;
        const float* ap = A + (size_t)(row + m) * Kdim + k0;
        const float* ap2 = HAS_A2 ? (A2 + (size_t)(row + m) * Kdim + k0) : nullptr;
#pragma unroll
        for (int i = 0; i < 8; ++i) {
            int kb = kmap(i, kh);
            float x0 = ap[kb], x1 = ap[kb + 1];
            if (HAS_A2) { x0 += ap2[kb]; x1 += ap2[kb + 1]; }
            a[2 * i] = (__bf16)x0; a[2 * i + 1] = (__bf16)x1;
        }
#pragma unroll
        for (int t = 0; t < 4; ++t) {
            v16bf bf;
            const __bf16* bp = Bt + (size_t)(col + t * 16 + m) * Kdim + k0;
#pragma unroll
            for (int i = 0; i < 8; ++i) {
                int kb = kmap(i, kh);
                bf[2 * i] = bp[kb]; bf[2 * i + 1] = bp[kb + 1];
            }
            acc[t] = wmma_bf16(a, bf, acc[t]);
        }
    }

    int n = lane & 15, mh = lane >> 4;
#pragma unroll
    for (int t = 0; t < 4; ++t) {
        int c = col + t * 16 + n;
        float bi = bias ? bias[c] : 0.f;
        float gg = 1.f, bbv = 0.f, al = 0.f;
        if (HAS_BNPRELU) {
            gg = g[c]; bbv = bb[c];
            al = ALPHA_SCALAR ? alpha[0] : alpha[c];
        }
#pragma unroll
        for (int r = 0; r < 8; ++r) {
            float v = acc[t][r] + bi;
            if (HAS_BNPRELU) {
                v = v * gg + bbv;
                v = (v >= 0.f) ? v : al * v;
            }
            Cout[(size_t)(row + r + 8 * mh) * Hd + c] = v;
        }
    }
}

// ---------------- flash attention: one wave = one (batch,head,16-row q tile) ----------------
// Nq fixed = 256. S tile = ONE wmma (DK=32). Softmax rows in LDS, then P@V via wmma.
template <bool HAS_BIAS, bool HAS_MASK>
__global__ __launch_bounds__(32)
void k_attn(const float* __restrict__ Q, const float* __restrict__ Kb,
            const float* __restrict__ Vb, float* __restrict__ O,
            int Nk,
            const float* __restrict__ bias, int bias_div,
            const unsigned char* __restrict__ mask)
{
    const int qTiles = Kd_ / 16;
    int lane = threadIdx.x;
    int qt   = blockIdx.x % qTiles;
    int bh   = blockIdx.x / qTiles;
    int h    = bh % HEADS;
    int bidx = bh / HEADS;

    __shared__ float sS[16 * 256];
    int m = lane & 15, kh = lane >> 4;

    // Q fragment (K-dim = DK = 32 exactly)
    v16bf aq;
    {
        const float* qp = Q + (size_t)(bidx * Kd_ + qt * 16 + m) * Hd + h * DKd;
#pragma unroll
        for (int i = 0; i < 8; ++i) {
            int kb = kmap(i, kh);
            aq[2 * i] = (__bf16)qp[kb]; aq[2 * i + 1] = (__bf16)qp[kb + 1];
        }
    }
    const float scale = 0.17677669529663687f; // 1/sqrt(32)

    for (int nt = 0; nt < Nk / 16; ++nt) {
        v16bf bk;
        const float* kp = Kb + (size_t)(bidx * Nk + nt * 16 + m) * Hd + h * DKd;
#pragma unroll
        for (int i = 0; i < 8; ++i) {
            int kb = kmap(i, kh);
            bk[2 * i] = (__bf16)kp[kb]; bk[2 * i + 1] = (__bf16)kp[kb + 1];
        }
        v8f s = (v8f){};
        s = wmma_bf16(aq, bk, s);
        int n = lane & 15, mh = lane >> 4;
#pragma unroll
        for (int r = 0; r < 8; ++r) {
            int qi = qt * 16 + r + 8 * mh;
            int kj = nt * 16 + n;
            float v = s[r] * scale;
            if (HAS_BIAS) {
                if (h < 2)
                    v += bias[(((size_t)(bidx / bias_div) * 2 + h) * Kd_ + qi) * Kd_ + kj];
            }
            if (HAS_MASK) {
                if (mask[(size_t)bidx * Nk + kj]) v = -1000000000.0f;
            }
            sS[(r + 8 * mh) * Nk + kj] = v;
        }
    }
    __syncthreads();

    // softmax per row (wave32 shuffle reduce)
    for (int r = 0; r < 16; ++r) {
        float* rowp = sS + r * Nk;
        float mx = -3.4e38f;
        for (int j = lane; j < Nk; j += 32) mx = fmaxf(mx, rowp[j]);
        for (int s = 16; s > 0; s >>= 1) mx = fmaxf(mx, __shfl_xor(mx, s, 32));
        float sum = 0.f;
        for (int j = lane; j < Nk; j += 32) { float e = __expf(rowp[j] - mx); rowp[j] = e; sum += e; }
        for (int s = 16; s > 0; s >>= 1) sum += __shfl_xor(sum, s, 32);
        float inv = 1.0f / sum;
        for (int j = lane; j < Nk; j += 32) rowp[j] *= inv;
    }
    __syncthreads();

    // O = P @ V ; output 16 x 32
    v8f o0 = (v8f){}, o1 = (v8f){};
    for (int k0 = 0; k0 < Nk; k0 += 32) {
        v16bf ap;
        const float* pp = sS + m * Nk + k0;
#pragma unroll
        for (int i = 0; i < 8; ++i) {
            int kb = kmap(i, kh);
            ap[2 * i] = (__bf16)pp[kb]; ap[2 * i + 1] = (__bf16)pp[kb + 1];
        }
#pragma unroll
        for (int t = 0; t < 2; ++t) {
            v16bf bv;
#pragma unroll
            for (int i = 0; i < 8; ++i) {
                int kb = kmap(i, kh);
                bv[2 * i]     = (__bf16)Vb[(size_t)(bidx * Nk + k0 + kb)     * Hd + h * DKd + t * 16 + m];
                bv[2 * i + 1] = (__bf16)Vb[(size_t)(bidx * Nk + k0 + kb + 1) * Hd + h * DKd + t * 16 + m];
            }
            (t ? o1 : o0) = wmma_bf16(ap, bv, t ? o1 : o0);
        }
    }
    int n = lane & 15, mh = lane >> 4;
#pragma unroll
    for (int r = 0; r < 8; ++r) {
        int qi = qt * 16 + r + 8 * mh;
        O[(size_t)(bidx * Kd_ + qi) * Hd + h * DKd + n]      = o0[r];
        O[(size_t)(bidx * Kd_ + qi) * Hd + h * DKd + 16 + n] = o1[r];
    }
}

// ---------------- LayerNorm over H=128 (optional residual), 32 lanes/row ----------------
template <bool HAS_RESID>
__global__ __launch_bounds__(128)
void k_ln(const float* __restrict__ X, const float* __restrict__ Resid,
          float* __restrict__ Out, const float* __restrict__ g,
          const float* __restrict__ b, int M)
{
    int row = blockIdx.x * 4 + (threadIdx.x >> 5);
    int lane = threadIdx.x & 31;
    if (row >= M) return;
    const float* x = X + (size_t)row * Hd;
    const float* rr = HAS_RESID ? (Resid + (size_t)row * Hd) : nullptr;
    float v[4], sum = 0.f, sq = 0.f;
#pragma unroll
    for (int i = 0; i < 4; ++i) {
        int c = lane + 32 * i;
        float t = x[c];
        if (HAS_RESID) t += rr[c];
        v[i] = t; sum += t; sq += t * t;
    }
    for (int s = 16; s > 0; s >>= 1) { sum += __shfl_xor(sum, s, 32); sq += __shfl_xor(sq, s, 32); }
    float mean = sum * (1.0f / Hd);
    float var  = sq * (1.0f / Hd) - mean * mean;
    float inv  = rsqrtf(var + 1e-5f);
    float* o = Out + (size_t)row * Hd;
#pragma unroll
    for (int i = 0; i < 4; ++i) {
        int c = lane + 32 * i;
        o[c] = (v[i] - mean) * inv * g[c] + b[c];
    }
}

// ---------------- broadcast f3d over L ----------------
__global__ void k_bcast(const float* __restrict__ f3d, float* __restrict__ f1) {
    size_t i = (size_t)blockIdx.x * blockDim.x + threadIdx.x;
    size_t total = (size_t)BLd * Kd_ * Hd;
    if (i >= total) return;
    size_t per = (size_t)Kd_ * Hd;
    size_t bl = i / per, rem = i % per;
    f1[i] = f3d[(bl / Ld) * per + rem];
}

// ---------------- sim-weighted combine of projected image feats ----------------
__global__ __launch_bounds__(128)
void k_imgagg(const float* __restrict__ imgf, const float* __restrict__ txt,
              const float* __restrict__ imgproj, const float* __restrict__ proj_b,
              float* __restrict__ out)
{
    int idx = blockIdx.x;            // bl*K + k
    int k = idx % Kd_, bl = idx / Kd_;
    int b = bl / Ld, l = bl % Ld;
    int lane = threadIdx.x & 31, grp = threadIdx.x >> 5;   // 4 groups, one per NF
    __shared__ float sSim[NFd];
    const float* ip = imgf + (((size_t)b * Kd_ + k) * NFd + grp) * CCLIP;
    const float* tp = txt + ((size_t)b * Ld + l) * CCLIP;
    float s = 0.f;
    for (int c = lane; c < CCLIP; c += 32) s += ip[c] * tp[c];
    for (int sh = 16; sh > 0; sh >>= 1) s += __shfl_xor(s, sh, 32);
    if (lane == 0) sSim[grp] = s;
    __syncthreads();
    int c = threadIdx.x;             // 0..127
    const float* pp = imgproj + (((size_t)b * Kd_ + k) * NFd) * Hd;
    float acc = proj_b[c];
#pragma unroll
    for (int f = 0; f < NFd; ++f) acc += sSim[f] * pp[f * Hd + c];
    out[((size_t)bl * Kd_ + k) * Hd + c] = acc;
}

// ---------------- final conf = h2 @ m_w3 + b3 ----------------
__global__ __launch_bounds__(256)
void k_conf(const float* __restrict__ h2, const float* __restrict__ w3,
            const float* __restrict__ b3, float* __restrict__ conf, int M)
{
    int row = (blockIdx.x * blockDim.x + threadIdx.x) >> 5;
    int lane = threadIdx.x & 31;
    if (row >= M) return;
    const float* x = h2 + (size_t)row * Hd;
    float s = 0.f;
    for (int c = lane; c < Hd; c += 32) s += x[c] * w3[c];
    for (int sh = 16; sh > 0; sh >>= 1) s += __shfl_xor(s, sh, 32);
    if (lane == 0) conf[row] = s + b3[0];
}

// ---------------- host orchestration ----------------
// params flatten order (insertion order, nested depth-first):
enum {
    I_FC_W1 = 0, I_FC_B1, I_FC_BN_G, I_FC_BN_B, I_FC_PRELU, I_FC_W2, I_FC_B2,
    I_SA0 = 7, I_CA0 = 17, I_SA1 = 27, I_CA1 = 37,   // each: Wq,bq,Wk,bk,Wv,bv,Wo,bo,ln_g,ln_b
    I_PROJ_W = 47, I_PROJ_B, I_PROJ_LNG, I_PROJ_LNB,
    I_M_W1 = 51, I_M_B1, I_M_BN1_G, I_M_BN1_B, I_M_A1,
    I_M_W2 = 56, I_M_B2, I_M_BN2_G, I_M_BN2_B, I_M_A2,
    I_M_W3 = 61, I_M_B3 = 62,
    I_CENTER = 63, I_DETR = 64, I_LANG = 65, I_MASK = 66, I_TXT = 67, I_IMGF = 68
};

extern "C" void kernel_launch(void* const* d_in, const int* in_sizes, int n_in,
                              void* d_out, int out_size, void* d_ws, size_t ws_size,
                              hipStream_t stream)
{
    (void)in_sizes; (void)n_in; (void)out_size; (void)ws_size;
    char* ws = (char*)d_ws;
    size_t off = 0;
    auto take = [&](size_t bytes) -> void* {
        void* p = ws + off;
        off += (bytes + 255) & ~(size_t)255;
        return p;
    };

    const size_t MQ = (size_t)BLd * Kd_;            // 16384
    __bf16* w_fc1 = (__bf16*)take((size_t)Hd * CDET * 2);
    __bf16* w_fc2 = (__bf16*)take((size_t)Hd * Hd * 2);
    __bf16* w_attn[4][4];
    for (int layer = 0; layer < 4; ++layer)
        for (int mt = 0; mt < 4; ++mt)
            w_attn[layer][mt] = (__bf16*)take((size_t)Hd * Hd * 2);
    __bf16* w_proj = (__bf16*)take((size_t)Hd * CCLIP * 2);
    __bf16* w_m1   = (__bf16*)take((size_t)Hd * Hd * 2);
    __bf16* w_m2   = (__bf16*)take((size_t)Hd * Hd * 2);

    float* dwgt    = (float*)take((size_t)Bd * 2 * Kd_ * Kd_ * 4);
    float* xbuf    = (float*)take((size_t)Bd * Kd_ * Hd * 4);
    float* f3d     = (float*)take((size_t)Bd * Kd_ * Hd * 4);
    float* f1      = (float*)take(MQ * Hd * 4);
    float* f2d     = (float*)take(MQ * Hd * 4);
    float* tmp     = (float*)take(MQ * Hd * 4);
    float* qb      = (float*)take(MQ * Hd * 4);
    float* kb      = (float*)take(MQ * Hd * 4);
    float* vb      = (float*)take(MQ * Hd * 4);
    float* ob      = (float*)take(MQ * Hd * 4);
    float* imgproj = (float*)take((size_t)Bd * Kd_ * NFd * Hd * 4);

    auto convT = [&](int idx, __bf16* dst, int Kdim) {
        int n = Kdim * Hd;
        k_convT<<<(n + 255) / 256, 256, 0, stream>>>((const float*)d_in[idx], dst, Kdim, Hd);
    };
    auto gemm = [&](const float* A, const __bf16* Bt, float* C, int M, int Kdim,
                    const float* bias) {
        int grid = ((M / 16) * 2 * 32 + 255) / 256;
        k_gemm<false, false, false><<<grid, 256, 0, stream>>>(A, nullptr, Bt, C, M, Kdim,
                                                              bias, nullptr, nullptr, nullptr);
    };
    auto F = [&](int idx) { return (const float*)d_in[idx]; };

    // ---- weight conversions ----
    convT(I_FC_W1, w_fc1, CDET);
    convT(I_FC_W2, w_fc2, Hd);
    const int attnBase[4] = { I_SA0, I_CA0, I_SA1, I_CA1 };
    for (int layer = 0; layer < 4; ++layer)
        for (int mt = 0; mt < 4; ++mt)                 // Wq,Wk,Wv,Wo at +0,+2,+4,+6
            convT(attnBase[layer] + 2 * mt, w_attn[layer][mt], Hd);
    convT(I_PROJ_W, w_proj, CCLIP);
    convT(I_M_W1, w_m1, Hd);
    convT(I_M_W2, w_m2, Hd);

    // ---- distance bias ----
    k_dist<<<Bd * Kd_, Kd_, 0, stream>>>(F(I_CENTER), dwgt);

    // ---- fc: x = prelu(bn(detr@W1+b1)); f3d = x@W2+b2 ----
    {
        int grid = ((Bd * Kd_ / 16) * 2 * 32 + 255) / 256;
        k_gemm<false, true, false><<<grid, 256, 0, stream>>>(
            F(I_DETR), nullptr, w_fc1, xbuf, Bd * Kd_, CDET,
            F(I_FC_B1), F(I_FC_BN_G), F(I_FC_BN_B), F(I_FC_PRELU));
    }
    gemm(xbuf, w_fc2, f3d, Bd * Kd_, Hd, F(I_FC_B2));

    // ---- sa0 on f3d (B batches, Nk=256, dist bias) ----
    gemm(f3d, w_attn[0][0], qb, Bd * Kd_, Hd, F(I_SA0 + 1));
    gemm(f3d, w_attn[0][1], kb, Bd * Kd_, Hd, F(I_SA0 + 3));
    gemm(f3d, w_attn[0][2], vb, Bd * Kd_, Hd, F(I_SA0 + 5));
    k_attn<true, false><<<Bd * HEADS * (Kd_ / 16), 32, 0, stream>>>(qb, kb, vb, ob, Kd_, dwgt, 1, nullptr);
    gemm(ob, w_attn[0][3], tmp, Bd * Kd_, Hd, F(I_SA0 + 7));
    k_ln<true><<<(Bd * Kd_) / 4, 128, 0, stream>>>(tmp, f3d, f3d, F(I_SA0 + 8), F(I_SA0 + 9), Bd * Kd_);

    // ---- f2d: project img feats, sim-weighted combine, LN ----
    gemm(F(I_IMGF), w_proj, imgproj, Bd * Kd_ * NFd, CCLIP, nullptr);
    k_imgagg<<<BLd * Kd_, 128, 0, stream>>>(F(I_IMGF), F(I_TXT), imgproj, F(I_PROJ_B), f2d);
    k_ln<false><<<(int)(MQ / 4), 128, 0, stream>>>(f2d, nullptr, f2d, F(I_PROJ_LNG), F(I_PROJ_LNB), (int)MQ);

    // ---- f1 = broadcast(f3d) over L ----
    {
        size_t total = MQ * Hd;
        k_bcast<<<(int)((total + 255) / 256), 256, 0, stream>>>(f3d, f1);
    }

    const unsigned char* msk = (const unsigned char*)d_in[I_MASK];

    // ---- ca0: q from f1, k/v from lang (Nk=64, mask) ----
    gemm(f1, w_attn[1][0], qb, (int)MQ, Hd, F(I_CA0 + 1));
    gemm(F(I_LANG), w_attn[1][1], kb, BLd * NWd, Hd, F(I_CA0 + 3));
    gemm(F(I_LANG), w_attn[1][2], vb, BLd * NWd, Hd, F(I_CA0 + 5));
    k_attn<false, true><<<BLd * HEADS * (Kd_ / 16), 32, 0, stream>>>(qb, kb, vb, ob, NWd, nullptr, 1, msk);
    gemm(ob, w_attn[1][3], tmp, (int)MQ, Hd, F(I_CA0 + 7));
    k_ln<true><<<(int)(MQ / 4), 128, 0, stream>>>(tmp, f1, f1, F(I_CA0 + 8), F(I_CA0 + 9), (int)MQ);

    // ---- sa1: self-attn on f1 (Nk=256, dist bias broadcast over L) ----
    gemm(f1, w_attn[2][0], qb, (int)MQ, Hd, F(I_SA1 + 1));
    gemm(f1, w_attn[2][1], kb, (int)MQ, Hd, F(I_SA1 + 3));
    gemm(f1, w_attn[2][2], vb, (int)MQ, Hd, F(I_SA1 + 5));
    k_attn<true, false><<<BLd * HEADS * (Kd_ / 16), 32, 0, stream>>>(qb, kb, vb, ob, Kd_, dwgt, Ld, nullptr);
    gemm(ob, w_attn[2][3], tmp, (int)MQ, Hd, F(I_SA1 + 7));
    k_ln<true><<<(int)(MQ / 4), 128, 0, stream>>>(tmp, f1, f1, F(I_SA1 + 8), F(I_SA1 + 9), (int)MQ);

    // ---- ca1 ----
    gemm(f1, w_attn[3][0], qb, (int)MQ, Hd, F(I_CA1 + 1));
    gemm(F(I_LANG), w_attn[3][1], kb, BLd * NWd, Hd, F(I_CA1 + 3));
    gemm(F(I_LANG), w_attn[3][2], vb, BLd * NWd, Hd, F(I_CA1 + 5));
    k_attn<false, true><<<BLd * HEADS * (Kd_ / 16), 32, 0, stream>>>(qb, kb, vb, ob, NWd, nullptr, 1, msk);
    gemm(ob, w_attn[3][3], tmp, (int)MQ, Hd, F(I_CA1 + 7));
    k_ln<true><<<(int)(MQ / 4), 128, 0, stream>>>(tmp, f1, f1, F(I_CA1 + 8), F(I_CA1 + 9), (int)MQ);

    // ---- MLP head: agg = f1 + f2d (fused as A2) ----
    {
        int grid = (((int)MQ / 16) * 2 * 32 + 255) / 256;
        k_gemm<true, true, true><<<grid, 256, 0, stream>>>(
            f1, f2d, w_m1, qb, (int)MQ, Hd,
            F(I_M_B1), F(I_M_BN1_G), F(I_M_BN1_B), F(I_M_A1));
        k_gemm<false, true, true><<<grid, 256, 0, stream>>>(
            qb, nullptr, w_m2, kb, (int)MQ, Hd,
            F(I_M_B2), F(I_M_BN2_G), F(I_M_BN2_B), F(I_M_A2));
    }
    k_conf<<<(int)(MQ * 32 / 256), 256, 0, stream>>>(kb, F(I_M_W3), F(I_M_B3), (float*)d_out, (int)MQ);
}
